// NTM_30159260352819
// MI455X (gfx1250) — compile-verified, matrix operand order checked by reading, not run on hardware
//
#include <hip/hip_runtime.h>
#include <math.h>
#include <stdint.h>

// ---------------------------------------------------------------------------
// NTM forward for MI455X (gfx1250): f16 WMMA GEMMs + fused addressing kernel.
// ---------------------------------------------------------------------------

typedef __attribute__((ext_vector_type(16))) _Float16 v16h;
typedef __attribute__((ext_vector_type(8)))  _Float16 v8h;
typedef __attribute__((ext_vector_type(8)))  float    v8f;

#define B_    128
#define T_    256
#define IN_   512
#define C_    2048
#define N_    256
#define M_    128
#define OUT_  512
#define KC    640      // M_ + IN_  (concat [r, x_t])
#define NCAT  1088     // 134 + 390 + 512 = 1036, padded to 17*64
#define EPSF  1e-8f

// ----------------------------- small helpers -------------------------------

__device__ __forceinline__ float sigmoidf_(float x) { return 1.f / (1.f + expf(-x)); }
__device__ __forceinline__ float softplusf_(float x) {
  return (x > 0.f) ? (x + log1pf(expf(-x))) : log1pf(expf(x));
}

__device__ __forceinline__ float wave_sum(float v) {
#pragma unroll
  for (int off = 16; off; off >>= 1) v += __shfl_xor(v, off, 32);
  return v;
}

__device__ __forceinline__ float block_sum(float v, volatile float* s8, int tid) {
  v = wave_sum(v);
  __syncthreads();                       // protect s8 from previous use
  if ((tid & 31) == 0) s8[tid >> 5] = v;
  __syncthreads();
  float r = 0.f;
#pragma unroll
  for (int i = 0; i < 8; i++) r += s8[i];
  return r;
}

__device__ __forceinline__ float block_max(float v, volatile float* s8, int tid) {
#pragma unroll
  for (int off = 16; off; off >>= 1) v = fmaxf(v, __shfl_xor(v, off, 32));
  __syncthreads();
  if ((tid & 31) == 0) s8[tid >> 5] = v;
  __syncthreads();
  float r = -3.4e38f;
#pragma unroll
  for (int i = 0; i < 8; i++) r = fmaxf(r, s8[i]);
  return r;
}

__device__ __forceinline__ v16h cat8(v8h lo, v8h hi) {
  return __builtin_shufflevector(lo, hi, 0, 1, 2, 3, 4, 5, 6, 7,
                                 8, 9, 10, 11, 12, 13, 14, 15);
}

// ----------------------------- prep kernels --------------------------------

// WcT[c, k] = (f16) Wc[k, c]   : C_ x KC, contiguous along K
__global__ void ntm_prep_wc(const float* __restrict__ Wc, _Float16* __restrict__ WcT) {
  int idx = blockIdx.x * 256 + threadIdx.x;
  if (idx >= C_ * KC) return;
  int c = idx / KC, k = idx - c * KC;
  WcT[idx] = (_Float16)Wc[(size_t)k * C_ + c];
}

// WcatT[j, c] = concat of Wr / Ww / Wout columns (transposed), zero padded.
__global__ void ntm_prep_cat(const float* __restrict__ Wr, const float* __restrict__ Ww,
                             const float* __restrict__ Wout,
                             const float* __restrict__ br, const float* __restrict__ bw,
                             const float* __restrict__ bout,
                             _Float16* __restrict__ WcatT, float* __restrict__ bcat) {
  int idx = blockIdx.x * 256 + threadIdx.x;
  if (idx >= NCAT * C_) return;
  int j = idx / C_, c = idx - j * C_;
  float v;
  if (j < 134)        v = Wr[(size_t)c * 134 + j];
  else if (j < 524)   v = Ww[(size_t)c * 390 + (j - 134)];
  else if (j < 1036)  v = Wout[(size_t)c * 512 + (j - 524)];
  else                v = 0.f;
  WcatT[idx] = (_Float16)v;
  if (c == 0) {
    bcat[j] = (j < 134) ? br[j] : (j < 524) ? bw[j - 134]
              : (j < 1036) ? bout[j - 524] : 0.f;
  }
}

// State init each call (ws is not re-poisoned between graph replays).
__global__ void ntm_init(const float* __restrict__ mem_init,
                         const float* __restrict__ read_init,
                         float* __restrict__ mem, float* __restrict__ wr,
                         float* __restrict__ ww, _Float16* __restrict__ Abuf) {
  int idx = blockIdx.x * 256 + threadIdx.x;
  if (idx < B_ * N_ * M_) mem[idx] = mem_init[idx % (N_ * M_)];
  if (idx < B_ * N_) { wr[idx] = 1.0f / N_; ww[idx] = 1.0f / N_; }
  if (idx < B_ * M_) {
    int b = idx / M_, m = idx - b * M_;
    Abuf[(size_t)b * KC + m] = (_Float16)read_init[m];
  }
}

// Pack x_t (f32 -> f16) into columns [128, 640) of Abuf.
__global__ void ntm_pack_x(const float* __restrict__ x, _Float16* __restrict__ Abuf, int t) {
  int idx = blockIdx.x * 256 + threadIdx.x;
  if (idx >= B_ * IN_) return;
  int b = idx >> 9, i = idx & 511;
  Abuf[(size_t)b * KC + M_ + i] = (_Float16)x[((size_t)b * T_ + t) * IN_ + i];
}

// ------------------------- WMMA GEMM 1: h = tanh(A Wc + bc) ----------------
// A: B_ x KC (f16), BT: C_ x KC (f16, row = output column), H: B_ x C_ (f16)
__global__ __launch_bounds__(256) void ntm_gemm_h(
    const _Float16* __restrict__ A, const _Float16* __restrict__ BT,
    const float* __restrict__ bias, _Float16* __restrict__ H) {
  const int wv = threadIdx.x >> 5, lane = threadIdx.x & 31;
  const int rowBase = blockIdx.y * 64 + (wv & 3) * 16;
  const int colBase = blockIdx.x * 64 + (wv >> 2) * 32;
  const int r = lane & 15, kh = lane >> 4;
  const _Float16* pA  = A  + (size_t)(rowBase + r) * KC + kh * 8;
  const _Float16* pB0 = BT + (size_t)(colBase + r) * KC + kh * 8;
  const _Float16* pB1 = pB0 + (size_t)16 * KC;
  v8f c0 = {}, c1 = {};
  for (int k0 = 0; k0 < KC; k0 += 32) {
    __builtin_prefetch(pB0 + 64, 0, 1);
    __builtin_prefetch(pB1 + 64, 0, 1);
    v8h alo = *(const v8h*)pA,  ahi = *(const v8h*)(pA + 16);
    v8h b0l = *(const v8h*)pB0, b0h = *(const v8h*)(pB0 + 16);
    v8h b1l = *(const v8h*)pB1, b1h = *(const v8h*)(pB1 + 16);
    v16h a  = cat8(alo, ahi);
    v16h b0 = cat8(b0l, b0h);
    v16h b1 = cat8(b1l, b1h);
    c0 = __builtin_amdgcn_wmma_f32_16x16x32_f16(false, a, false, b0, (short)0, c0, false, false);
    c1 = __builtin_amdgcn_wmma_f32_16x16x32_f16(false, a, false, b1, (short)0, c1, false, false);
    pA += 32; pB0 += 32; pB1 += 32;
  }
  const int col = colBase + r;
#pragma unroll
  for (int i = 0; i < 8; i++) {
    int grow = rowBase + kh * 8 + i;
    H[(size_t)grow * C_ + col]      = (_Float16)tanhf(c0[i] + bias[col]);
    H[(size_t)grow * C_ + col + 16] = (_Float16)tanhf(c1[i] + bias[col + 16]);
  }
}

// --------- WMMA GEMM 2: G = H [Wr|Ww|Wout] + bcat ; scatter Wout slice -----
__global__ __launch_bounds__(256) void ntm_gemm_proj(
    const _Float16* __restrict__ H, const _Float16* __restrict__ BT,
    const float* __restrict__ bcat, float* __restrict__ Gout,
    float* __restrict__ Out, int t) {
  const int wv = threadIdx.x >> 5, lane = threadIdx.x & 31;
  const int rowBase = blockIdx.y * 64 + (wv & 3) * 16;
  const int colBase = blockIdx.x * 64 + (wv >> 2) * 32;
  const int r = lane & 15, kh = lane >> 4;
  const _Float16* pA  = H  + (size_t)(rowBase + r) * C_ + kh * 8;
  const _Float16* pB0 = BT + (size_t)(colBase + r) * C_ + kh * 8;
  const _Float16* pB1 = pB0 + (size_t)16 * C_;
  v8f c0 = {}, c1 = {};
  for (int k0 = 0; k0 < C_; k0 += 32) {
    __builtin_prefetch(pB0 + 64, 0, 1);
    __builtin_prefetch(pB1 + 64, 0, 1);
    v8h alo = *(const v8h*)pA,  ahi = *(const v8h*)(pA + 16);
    v8h b0l = *(const v8h*)pB0, b0h = *(const v8h*)(pB0 + 16);
    v8h b1l = *(const v8h*)pB1, b1h = *(const v8h*)(pB1 + 16);
    v16h a  = cat8(alo, ahi);
    v16h b0 = cat8(b0l, b0h);
    v16h b1 = cat8(b1l, b1h);
    c0 = __builtin_amdgcn_wmma_f32_16x16x32_f16(false, a, false, b0, (short)0, c0, false, false);
    c1 = __builtin_amdgcn_wmma_f32_16x16x32_f16(false, a, false, b1, (short)0, c1, false, false);
    pA += 32; pB0 += 32; pB1 += 32;
  }
  const int col = colBase + r;
#pragma unroll
  for (int i = 0; i < 8; i++) {
    int grow = rowBase + kh * 8 + i;
    float v0 = c0[i] + bcat[col];
    float v1 = c1[i] + bcat[col + 16];
    Gout[(size_t)grow * NCAT + col]      = v0;
    Gout[(size_t)grow * NCAT + col + 16] = v1;
    if (col >= 524 && col < 1036)
      Out[((size_t)grow * T_ + t) * OUT_ + (col - 524)] = v0;
    int c1i = col + 16;
    if (c1i >= 524 && c1i < 1036)
      Out[((size_t)grow * T_ + t) * OUT_ + (c1i - 524)] = v1;
  }
}

// ---------------- fused addressing (read+write) + memory update ------------
// One block per batch element; 8 waves * 32 rows; lane owns m = 4*lane..+3.
// Layout of G row: [0,128)=k_r 128=beta 129=g 130..132=s 133=gamma |
//                  134+: same for write head | 268+: erase | 396+: add
__global__ __launch_bounds__(256) void ntm_addr(
    const float* __restrict__ G, float* __restrict__ mem,
    float* __restrict__ wr, float* __restrict__ ww,
    _Float16* __restrict__ Abuf) {
  __shared__ float s_simr[N_], s_simw[N_], s_wgr[N_], s_wgw[N_];
  __shared__ float s_red[8];
  __shared__ float s_racc[8 * M_];

  const int b = blockIdx.x, tid = threadIdx.x;
  const int wv = tid >> 5, lane = tid & 31;
  const float* g  = G + (size_t)b * NCAT;
  const float* pw = g + 134;

  // head parameters (uniform; broadcast from L2)
  float betar = softplusf_(g[128]);
  float gr    = sigmoidf_(g[129]);
  float q0 = g[130], q1 = g[131], q2 = g[132];
  float qm = fmaxf(q0, fmaxf(q1, q2));
  float qe0 = expf(q0 - qm), qe1 = expf(q1 - qm), qe2 = expf(q2 - qm);
  float qs = qe0 + qe1 + qe2;
  float sr0 = qe0 / qs, sr1 = qe1 / qs, sr2 = qe2 / qs;
  float gammar = 1.f + softplusf_(g[133]);

  float betaw = softplusf_(pw[128]);
  float gw    = sigmoidf_(pw[129]);
  float u0 = pw[130], u1 = pw[131], u2 = pw[132];
  float um = fmaxf(u0, fmaxf(u1, u2));
  float ue0 = expf(u0 - um), ue1 = expf(u1 - um), ue2 = expf(u2 - um);
  float us = ue0 + ue1 + ue2;
  float sw0 = ue0 / us, sw1 = ue1 / us, sw2 = ue2 / us;
  float gammaw = 1.f + softplusf_(pw[133]);

  // per-lane 4-element slice of the keys
  const int m0 = lane * 4;
  float kr0 = g[m0 + 0], kr1 = g[m0 + 1], kr2 = g[m0 + 2], kr3 = g[m0 + 3];
  float kw0 = pw[m0 + 0], kw1 = pw[m0 + 1], kw2 = pw[m0 + 2], kw3 = pw[m0 + 3];
  float normkr = sqrtf(wave_sum(kr0 * kr0 + kr1 * kr1 + kr2 * kr2 + kr3 * kr3));
  float normkw = sqrtf(wave_sum(kw0 * kw0 + kw1 * kw1 + kw2 * kw2 + kw3 * kw3));

  // pass 1: cosine similarities against all memory rows
  for (int i = 0; i < 32; i++) {
    int n = wv * 32 + i;
    const float* row = mem + ((size_t)b * N_ + n) * M_ + m0;
    float4 mv = *(const float4*)row;
    float dr = mv.x * kr0 + mv.y * kr1 + mv.z * kr2 + mv.w * kr3;
    float dw = mv.x * kw0 + mv.y * kw1 + mv.z * kw2 + mv.w * kw3;
    float nn = mv.x * mv.x + mv.y * mv.y + mv.z * mv.z + mv.w * mv.w;
    dr = wave_sum(dr); dw = wave_sum(dw); nn = wave_sum(nn);
    if (lane == 0) {
      float nr = sqrtf(nn);
      s_simr[n] = dr / (normkr * nr + EPSF);
      s_simw[n] = dw / (normkw * nr + EPSF);
    }
  }
  __syncthreads();

  // softmax(beta*sim), interpolate, shift, sharpen, renormalize (n = tid)
  float xr = betar * s_simr[tid];
  float xw = betaw * s_simw[tid];
  float mr = block_max(xr, s_red, tid);
  float mw = block_max(xw, s_red, tid);
  float er = expf(xr - mr), ew = expf(xw - mw);
  float sumr = block_sum(er, s_red, tid);
  float sumw = block_sum(ew, s_red, tid);
  float wgr_ = gr * (er / sumr) + (1.f - gr) * wr[(size_t)b * N_ + tid];
  float wgw_ = gw * (ew / sumw) + (1.f - gw) * ww[(size_t)b * N_ + tid];
  s_wgr[tid] = wgr_; s_wgw[tid] = wgw_;
  __syncthreads();
  float wtr = sr0 * s_wgr[(tid + 1) & (N_ - 1)] + sr1 * wgr_ +
              sr2 * s_wgr[(tid + N_ - 1) & (N_ - 1)];
  float wtw = sw0 * s_wgw[(tid + 1) & (N_ - 1)] + sw1 * wgw_ +
              sw2 * s_wgw[(tid + N_ - 1) & (N_ - 1)];
  float wpr = exp2f(gammar * log2f(wtr + EPSF));
  float wpw = exp2f(gammaw * log2f(wtw + EPSF));
  float spr = block_sum(wpr, s_red, tid);
  float spw = block_sum(wpw, s_red, tid);
  float wrn = wpr / spr, wwn = wpw / spw;
  wr[(size_t)b * N_ + tid] = wrn;
  ww[(size_t)b * N_ + tid] = wwn;
  __syncthreads();
  s_wgr[tid] = wrn; s_wgw[tid] = wwn;
  __syncthreads();

  // pass 2: r_new = w_r . mem  (fused with erase/add memory update)
  float e0 = sigmoidf_(g[268 + m0 + 0]);
  float e1 = sigmoidf_(g[268 + m0 + 1]);
  float e2 = sigmoidf_(g[268 + m0 + 2]);
  float e3 = sigmoidf_(g[268 + m0 + 3]);
  float a0 = tanhf(g[396 + m0 + 0]);
  float a1 = tanhf(g[396 + m0 + 1]);
  float a2 = tanhf(g[396 + m0 + 2]);
  float a3 = tanhf(g[396 + m0 + 3]);
  float rc0 = 0.f, rc1 = 0.f, rc2 = 0.f, rc3 = 0.f;
  for (int i = 0; i < 32; i++) {
    int n = wv * 32 + i;
    float* row = mem + ((size_t)b * N_ + n) * M_ + m0;
    float4 mv = *(const float4*)row;
    float wn = s_wgr[n], vn = s_wgw[n];
    rc0 += wn * mv.x; rc1 += wn * mv.y; rc2 += wn * mv.z; rc3 += wn * mv.w;
    mv.x = mv.x * (1.f - vn * e0) + vn * a0;
    mv.y = mv.y * (1.f - vn * e1) + vn * a1;
    mv.z = mv.z * (1.f - vn * e2) + vn * a2;
    mv.w = mv.w * (1.f - vn * e3) + vn * a3;
    *(float4*)row = mv;
  }
  s_racc[wv * M_ + m0 + 0] = rc0;
  s_racc[wv * M_ + m0 + 1] = rc1;
  s_racc[wv * M_ + m0 + 2] = rc2;
  s_racc[wv * M_ + m0 + 3] = rc3;
  __syncthreads();
  if (tid < M_) {
    float rsum = 0.f;
#pragma unroll
    for (int w8 = 0; w8 < 8; w8++) rsum += s_racc[w8 * M_ + tid];
    Abuf[(size_t)b * KC + tid] = (_Float16)rsum;  // r part of next step's A
  }
}

// ------------------------------- host side ---------------------------------

extern "C" void kernel_launch(void* const* d_in, const int* in_sizes, int n_in,
                              void* d_out, int out_size, void* d_ws, size_t ws_size,
                              hipStream_t stream) {
  (void)in_sizes; (void)n_in; (void)out_size; (void)ws_size;
  const float* x         = (const float*)d_in[0];
  const float* Wc        = (const float*)d_in[1];
  const float* bc        = (const float*)d_in[2];
  const float* Wout      = (const float*)d_in[3];
  const float* bout      = (const float*)d_in[4];
  const float* Wr        = (const float*)d_in[5];
  const float* br        = (const float*)d_in[6];
  const float* Ww        = (const float*)d_in[7];
  const float* bw        = (const float*)d_in[8];
  const float* mem_init  = (const float*)d_in[9];
  const float* read_init = (const float*)d_in[10];
  float* Out = (float*)d_out;

  char* p = (char*)d_ws;
  auto carve = [&](size_t bytes) -> char* {
    uintptr_t q = ((uintptr_t)p + 255) & ~(uintptr_t)255;
    char* r = (char*)q;
    p = r + bytes;
    return r;
  };
  _Float16* WcT   = (_Float16*)carve((size_t)C_ * KC * sizeof(_Float16));
  _Float16* WcatT = (_Float16*)carve((size_t)NCAT * C_ * sizeof(_Float16));
  float*    bcat  = (float*)carve((size_t)NCAT * sizeof(float));
  _Float16* Abuf  = (_Float16*)carve((size_t)B_ * KC * sizeof(_Float16));
  _Float16* Hbf   = (_Float16*)carve((size_t)B_ * C_ * sizeof(_Float16));
  float*    GoutB = (float*)carve((size_t)B_ * NCAT * sizeof(float));
  float*    memS  = (float*)carve((size_t)B_ * N_ * M_ * sizeof(float));
  float*    wrS   = (float*)carve((size_t)B_ * N_ * sizeof(float));
  float*    wwS   = (float*)carve((size_t)B_ * N_ * sizeof(float));

  ntm_prep_wc<<<(C_ * KC + 255) / 256, 256, 0, stream>>>(Wc, WcT);
  ntm_prep_cat<<<(NCAT * C_ + 255) / 256, 256, 0, stream>>>(Wr, Ww, Wout, br, bw, bout,
                                                            WcatT, bcat);
  ntm_init<<<(B_ * N_ * M_ + 255) / 256, 256, 0, stream>>>(mem_init, read_init, memS,
                                                           wrS, wwS, Abuf);

  for (int t = 0; t < T_; ++t) {
    ntm_pack_x<<<(B_ * IN_ + 255) / 256, 256, 0, stream>>>(x, Abuf, t);
    ntm_gemm_h<<<dim3(C_ / 64, B_ / 64), 256, 0, stream>>>(Abuf, WcT, bc, Hbf);
    ntm_gemm_proj<<<dim3(NCAT / 64, B_ / 64), 256, 0, stream>>>(Hbf, WcatT, bcat, GoutB,
                                                                Out, t);
    ntm_addr<<<B_, 256, 0, stream>>>(GoutB, memS, wrS, wwS, Abuf);
  }
}
